// DualViewGNN_36498632081483
// MI455X (gfx1250) — compile-verified, matrix operand order checked by reading, not run on hardware
//
#include <hip/hip_runtime.h>
#include <stdint.h>

#define D_DIM 768
#define R_REL 8
#define LRELU 0.2f

typedef __attribute__((ext_vector_type(16))) __bf16 v16bf;
typedef __attribute__((ext_vector_type(8)))  float  v8f;

__device__ __forceinline__ __bf16 bfc(unsigned short u) {
  return __builtin_bit_cast(__bf16, u);
}
__device__ __forceinline__ unsigned short f2bf(float f) {
  unsigned u = __float_as_uint(f);
  u += 0x7FFFu + ((u >> 16) & 1u);
  return (unsigned short)(u >> 16);
}
__device__ __forceinline__ unsigned fenc(float f) {
  unsigned u = __float_as_uint(f);
  return (u & 0x80000000u) ? ~u : (u | 0x80000000u);
}
__device__ __forceinline__ float fdec(unsigned u) {
  return __uint_as_float((u & 0x80000000u) ? (u & 0x7FFFFFFFu) : ~u);
}
__device__ __forceinline__ float lrelu(float z) { return z > 0.f ? z : LRELU * z; }
__device__ __forceinline__ float eluf(float z)  { return z > 0.f ? z : (__expf(z) - 1.f); }

// Async Global->LDS DMA (CDNA5 GLOBAL_LOAD_ASYNC_TO_LDS_B128, tracked by ASYNCcnt).
// lds_off: byte offset of destination in LDS; gaddr: 16B-aligned global address.
__device__ __forceinline__ void async_b128(void* lds_dst, const void* gsrc) {
  unsigned loff = (unsigned)(size_t)lds_dst;
  unsigned long long ga = (unsigned long long)gsrc;
  asm volatile("global_load_async_to_lds_b128 %0, %1, off"
               :: "v"(loff), "v"(ga)
               : "memory");
}
__device__ __forceinline__ void wait_async0() {
  asm volatile("s_wait_asynccnt 0x0" ::: "memory");
}

__device__ __forceinline__ float block_reduce256(float v, float* red) {
  int tid = threadIdx.x;
  red[tid] = v; __syncthreads();
  for (int s = 128; s > 0; s >>= 1) {
    if (tid < s) red[tid] += red[tid + s];
    __syncthreads();
  }
  float r = red[0];
  __syncthreads();
  return r;
}

// ---------------- tiny utility kernels ----------------
__global__ void k_zero32(unsigned* __restrict__ p, size_t n) {
  size_t i = (size_t)blockIdx.x * 256 + threadIdx.x;
  if (i < n) p[i] = 0u;
}
__global__ void k_f2bf16(const float* __restrict__ in, unsigned short* __restrict__ out, size_t n) {
  size_t i = (size_t)blockIdx.x * 256 + threadIdx.x;
  if (i < n) out[i] = f2bf(in[i]);
}
// out[row] = dot(W[row,:], a)   (W: nrows x D row-major), one block per row
__global__ void k_matvec(const float* __restrict__ W, const float* __restrict__ a,
                         float* __restrict__ out, int D_) {
  __shared__ float red[256];
  int row = blockIdx.x;
  const float* wr = W + (size_t)row * D_;
  float s = 0.f;
  for (int c = threadIdx.x; c < D_; c += 256) s += wr[c] * a[c];
  float r = block_reduce256(s, red);
  if (threadIdx.x == 0) out[row] = r;
}
// out[n*stride] = dot(X[n,:], w) + bias  (one block per row n)
__global__ void k_rowdot(const float* __restrict__ X, int ldx,
                         const float* __restrict__ w, const float* __restrict__ biasPtr,
                         float* __restrict__ out, int outStride, int D_) {
  __shared__ float red[256];
  int n = blockIdx.x;
  const float* xr = X + (size_t)n * ldx;
  float s = 0.f;
  for (int c = threadIdx.x; c < D_; c += 256) s += xr[c] * w[c];
  float r = block_reduce256(s, red);
  if (threadIdx.x == 0) out[(size_t)n * outStride] = r + (biasPtr ? biasPtr[0] : 0.f);
}
// single dot product
__global__ void k_dot(const float* __restrict__ u, const float* __restrict__ v,
                      float* __restrict__ out, int n) {
  __shared__ float red[256];
  float s = 0.f;
  for (int i = threadIdx.x; i < n; i += 256) s += u[i] * v[i];
  float r = block_reduce256(s, red);
  if (threadIdx.x == 0) out[0] = r;
}
// bucket edges by relation (capacity cap per relation)
__global__ void k_bucket(const int* __restrict__ etype, int* __restrict__ cnt,
                         int* __restrict__ buckets, int E_, int cap) {
  int j = blockIdx.x * 256 + threadIdx.x;
  if (j < E_) {
    int r = etype[j];
    int p = atomicAdd(&cnt[r], 1);
    buckets[(size_t)r * cap + p] = j;
  }
}

// ---------------- segmented softmax over edges ----------------
__global__ void k_edge_logits(const int* __restrict__ seg, const int* __restrict__ oth,
                              const int* __restrict__ etype, int R_,
                              const float* __restrict__ sseg, const float* __restrict__ soth,
                              const float* __restrict__ baPtr,
                              float* __restrict__ e, unsigned* __restrict__ segmax, int E_) {
  int j = blockIdx.x * 256 + threadIdx.x;
  if (j < E_) {
    int r = etype ? etype[j] : 0;
    float ba = baPtr ? baPtr[0] : 0.f;
    size_t idx = (size_t)seg[j] * R_ + r;
    float l = lrelu(sseg[idx] + soth[(size_t)oth[j] * R_ + r] + ba);
    e[j] = l;
    atomicMax(&segmax[idx], fenc(l));
  }
}
__global__ void k_edge_exp(const int* __restrict__ seg, const int* __restrict__ etype, int R_,
                           float* __restrict__ e, const unsigned* __restrict__ segmax,
                           float* __restrict__ segsum, int E_) {
  int j = blockIdx.x * 256 + threadIdx.x;
  if (j < E_) {
    int r = etype ? etype[j] : 0;
    size_t idx = (size_t)seg[j] * R_ + r;
    float ex = __expf(e[j] - fdec(segmax[idx]));
    e[j] = ex;
    atomicAdd(&segsum[idx], ex);
  }
}
__global__ void k_edge_att(const int* __restrict__ seg, const int* __restrict__ etype, int R_,
                           float* __restrict__ e, const float* __restrict__ segsum, int E_) {
  int j = blockIdx.x * 256 + threadIdx.x;
  if (j < E_) {
    int r = etype ? etype[j] : 0;
    e[j] = e[j] / (segsum[(size_t)seg[j] * R_ + r] + 1e-10f);
  }
}

// ---------------- WMMA GEMM:  C[N x 768] = A[N x K](bf16) @ B[K x 768](bf16) + bias ----------------
// block 256 = 8 waves; each wave: one 16x16 tile; block tile 16 x 128.
// A (16x32) and B (32x128) K-step tiles staged via async Global->LDS B128 DMA.
__global__ void k_gemm(const unsigned short* __restrict__ A, int lda,
                       const unsigned short* __restrict__ B, int ldb,
                       const float* __restrict__ bias,
                       float* __restrict__ C, unsigned short* __restrict__ Cb, int ldcb,
                       int K) {
  __shared__ __align__(16) unsigned short ldsA[16][32];
  __shared__ __align__(16) unsigned short ldsB[32][128];
  int tid = threadIdx.x, lane = tid & 31, wave = tid >> 5;
  int m0 = blockIdx.x * 16;
  int nb = blockIdx.y * 128;
  int arow = lane & 15, kh = (lane >> 4) << 3;
  int bcol = wave * 16 + (lane & 15);
  v8f acc = {0.f, 0.f, 0.f, 0.f, 0.f, 0.f, 0.f, 0.f};
  for (int k0 = 0; k0 < K; k0 += 32) {
    __syncthreads();
    if (tid < 64) {  // waves 0-1 exactly: wave-uniform issue
      int rr = tid >> 2, cc = (tid & 3) * 8;
      async_b128(&ldsA[rr][cc], A + (size_t)(m0 + rr) * lda + k0 + cc);
    }
    for (int q = tid; q < 512; q += 256) {  // 2 issues per thread
      int rr = q >> 4, cc = (q & 15) * 8;
      async_b128(&ldsB[rr][cc], B + (size_t)(k0 + rr) * ldb + nb + cc);
    }
    if (k0 + 32 < K) __builtin_prefetch(B + (size_t)(k0 + 32) * ldb + nb + bcol * 2, 0, 1);
    wait_async0();
    __syncthreads();
    v16bf af, bf;
#pragma unroll
    for (int i = 0; i < 8; ++i) {
      int kk = ((i >> 2) << 4) + kh + ((i & 3) << 1);
      af[2 * i]     = bfc(ldsA[arow][kk]);
      af[2 * i + 1] = bfc(ldsA[arow][kk + 1]);
      bf[2 * i]     = bfc(ldsB[kk][bcol]);
      bf[2 * i + 1] = bfc(ldsB[kk + 1][bcol]);
    }
    acc = __builtin_amdgcn_wmma_f32_16x16x32_bf16(false, af, false, bf, (short)0, acc, false, false);
  }
  int ccol = lane & 15, crow = (lane >> 4) << 3;
  int col = nb + wave * 16 + ccol;
  float bb = bias ? bias[col] : 0.f;
#pragma unroll
  for (int j = 0; j < 8; ++j) {
    int row = m0 + crow + j;
    float v = acc[j] + bb;
    if (C)  C[(size_t)row * D_DIM + col] = v;
    if (Cb) Cb[(size_t)row * ldcb + col] = f2bf(v);
  }
}

// ---------------- RGAT aggregate: gather 16 edges of relation r, V = x[src] @ WV[r], h[tgt] += att*V ----------------
__global__ void k_rgat_agg(const unsigned short* __restrict__ xb,
                           const unsigned short* __restrict__ WVb,
                           const int* __restrict__ buckets, const int* __restrict__ cnt,
                           const int* __restrict__ src, const int* __restrict__ tgt,
                           const float* __restrict__ att, float* __restrict__ h, int cap) {
  int r = blockIdx.z;
  int t0 = blockIdx.x * 16;
  int c = cnt[r];
  if (t0 >= c) return;  // uniform per block
  __shared__ int   eRow[16];
  __shared__ int   eTgt[16];
  __shared__ float eAtt[16];
  __shared__ __align__(16) unsigned short ldsA[16][32];
  __shared__ __align__(16) unsigned short ldsB[32][128];
  int tid = threadIdx.x;
  if (tid < 16) {
    int idx = t0 + tid;
    if (idx < c) {
      int ej = buckets[(size_t)r * cap + idx];
      eRow[tid] = src[ej]; eTgt[tid] = tgt[ej]; eAtt[tid] = att[ej];
    } else {
      eRow[tid] = 0; eTgt[tid] = 0; eAtt[tid] = 0.f;  // padded rows contribute 0
    }
  }
  __syncthreads();
  int lane = tid & 31, wave = tid >> 5;
  int nb = blockIdx.y * 128;
  const unsigned short* B = WVb + (size_t)r * D_DIM * D_DIM;
  int arow = lane & 15, kh = (lane >> 4) << 3;
  int bcol = wave * 16 + (lane & 15);
  v8f acc = {0.f, 0.f, 0.f, 0.f, 0.f, 0.f, 0.f, 0.f};
  for (int k0 = 0; k0 < D_DIM; k0 += 32) {
    __syncthreads();
    if (tid < 64) {  // gathered rows
      int rr = tid >> 2, cc = (tid & 3) * 8;
      async_b128(&ldsA[rr][cc], xb + (size_t)eRow[rr] * D_DIM + k0 + cc);
    }
    for (int q = tid; q < 512; q += 256) {
      int rr = q >> 4, cc = (q & 15) * 8;
      async_b128(&ldsB[rr][cc], B + (size_t)(k0 + rr) * D_DIM + nb + cc);
    }
    if (k0 + 32 < D_DIM) __builtin_prefetch(B + (size_t)(k0 + 32) * D_DIM + nb + bcol * 2, 0, 1);
    wait_async0();
    __syncthreads();
    v16bf af, bf;
#pragma unroll
    for (int i = 0; i < 8; ++i) {
      int kk = ((i >> 2) << 4) + kh + ((i & 3) << 1);
      af[2 * i]     = bfc(ldsA[arow][kk]);
      af[2 * i + 1] = bfc(ldsA[arow][kk + 1]);
      bf[2 * i]     = bfc(ldsB[kk][bcol]);
      bf[2 * i + 1] = bfc(ldsB[kk + 1][bcol]);
    }
    acc = __builtin_amdgcn_wmma_f32_16x16x32_bf16(false, af, false, bf, (short)0, acc, false, false);
  }
  int ccol = lane & 15, crow = (lane >> 4) << 3;
#pragma unroll
  for (int j = 0; j < 8; ++j) {
    int m = crow + j;
    float v = acc[j] * eAtt[m];
    atomicAdd(&h[(size_t)eTgt[m] * D_DIM + nb + wave * 16 + ccol], v);
  }
}

// out[seg[e],:] += att[e] * V[gat[e],:]
__global__ void k_scatter(const int* __restrict__ seg, const int* __restrict__ gat,
                          const float* __restrict__ att, const float* __restrict__ V,
                          float* __restrict__ out) {
  int e = blockIdx.x;
  float a = att[e];
  int s = seg[e], g = gat[e];
  for (int c = threadIdx.x; c < D_DIM; c += 256)
    atomicAdd(&out[(size_t)s * D_DIM + c], a * V[(size_t)g * D_DIM + c]);
}

// out = LN(base + maybe_elu(add)); optional f32 and bf16 outputs
__global__ void k_ln(const float* __restrict__ base, const float* __restrict__ add, int do_elu,
                     const float* __restrict__ g, const float* __restrict__ b,
                     float* __restrict__ outF, unsigned short* __restrict__ outB,
                     int ldOutB, int offOutB) {
  __shared__ float vals[D_DIM];
  __shared__ float red[256];
  int n = blockIdx.x, tid = threadIdx.x;
  float s = 0.f;
  for (int c = tid; c < D_DIM; c += 256) {
    float v = add[(size_t)n * D_DIM + c];
    if (do_elu) v = eluf(v);
    v += base[(size_t)n * D_DIM + c];
    vals[c] = v; s += v;
  }
  float mean = block_reduce256(s, red) * (1.f / D_DIM);
  s = 0.f;
  for (int c = tid; c < D_DIM; c += 256) { float d = vals[c] - mean; s += d * d; }
  float var = block_reduce256(s, red) * (1.f / D_DIM);
  float rstd = rsqrtf(var + 1e-5f);
  for (int c = tid; c < D_DIM; c += 256) {
    float o = (vals[c] - mean) * rstd * g[c] + b[c];
    if (outF) outF[(size_t)n * D_DIM + c] = o;
    if (outB) outB[(size_t)n * ldOutB + offOutB + c] = f2bf(o);
  }
}

extern "C" void kernel_launch(void* const* d_in, const int* in_sizes, int n_in,
                              void* d_out, int out_size, void* d_ws, size_t ws_size,
                              hipStream_t stream) {
  (void)n_in; (void)out_size; (void)ws_size;
  const int N = in_sizes[0] / D_DIM;   // 20000
  const int E = in_sizes[2];           // 100000

  const float* x    = (const float*)d_in[0];
  const int*   ei   = (const int*)d_in[1];
  const int*   et   = (const int*)d_in[2];
  const int*   cei  = (const int*)d_in[3];
  const int*   dei  = (const int*)d_in[4];
  const float* WQf  = (const float*)d_in[5];
  const float* WKf  = (const float*)d_in[6];
  const float* WVf  = (const float*)d_in[7];
  const float* af   = (const float*)d_in[8];
  const float* WQi  = (const float*)d_in[9];
  const float* WKi  = (const float*)d_in[10];
  const float* WVi  = (const float*)d_in[11];
  const float* ai   = (const float*)d_in[12];
  const float* Wpf  = (const float*)d_in[13];
  const float* bpf  = (const float*)d_in[14];
  const float* waf  = (const float*)d_in[15];
  const float* baf  = (const float*)d_in[16];
  const float* Wpi  = (const float*)d_in[17];
  const float* bpi  = (const float*)d_in[18];
  const float* wai  = (const float*)d_in[19];
  const float* bai  = (const float*)d_in[20];
  const float* gf1  = (const float*)d_in[21];
  const float* bf1  = (const float*)d_in[22];
  const float* gi1  = (const float*)d_in[23];
  const float* bi1  = (const float*)d_in[24];
  const float* gf2  = (const float*)d_in[25];
  const float* bf2  = (const float*)d_in[26];
  const float* gi2  = (const float*)d_in[27];
  const float* bi2  = (const float*)d_in[28];
  const float* Wpa  = (const float*)d_in[29];
  const float* bpa  = (const float*)d_in[30];
  const float* Wdoc = (const float*)d_in[31];
  const float* adoc = (const float*)d_in[32];
  const float* gdoc = (const float*)d_in[33];
  const float* bdoc = (const float*)d_in[34];

  const int* src = ei;        const int* tgt = ei + E;
  const int* qi  = cei;       const int* ki  = cei + E;
  const int* dsr = dei;       const int* dtg = dei + E;

  // ---- workspace bump allocator ----
  char* wsb = (char*)d_ws;
  size_t off = 0;
  auto alloc = [&](size_t bytes) -> void* {
    void* p = wsb + off;
    off = (off + bytes + 255) & ~(size_t)255;
    return p;
  };
  const size_t ND = (size_t)N * D_DIM;
  unsigned short* XB     = (unsigned short*)alloc(ND * 2);
  unsigned short* WVFB   = (unsigned short*)alloc((size_t)R_REL * D_DIM * D_DIM * 2);
  unsigned short* WVIB   = (unsigned short*)alloc((size_t)R_REL * D_DIM * D_DIM * 2);
  unsigned short* WPFB   = (unsigned short*)alloc((size_t)D_DIM * D_DIM * 2);
  unsigned short* WPIB   = (unsigned short*)alloc((size_t)D_DIM * D_DIM * 2);
  unsigned short* WPARAB = (unsigned short*)alloc((size_t)2 * D_DIM * D_DIM * 2);
  unsigned short* WDOCB  = (unsigned short*)alloc((size_t)D_DIM * D_DIM * 2);
  float* WQv   = (float*)alloc((size_t)R_REL * D_DIM * 4);
  float* WKv   = (float*)alloc((size_t)R_REL * D_DIM * 4);
  float* SQ    = (float*)alloc((size_t)N * R_REL * 4);
  float* SK    = (float*)alloc((size_t)N * R_REL * 4);
  float* EATT  = (float*)alloc((size_t)E * 4);
  unsigned* SEGMAX = (unsigned*)alloc((size_t)N * R_REL * 4);
  float*    SEGSUM = (float*)alloc((size_t)N * R_REL * 4);
  int* CNT  = (int*)alloc(R_REL * 4);
  int* BUCK = (int*)alloc((size_t)R_REL * E * 4);
  float* ACC  = (float*)alloc(ND * 4);
  float* HF   = (float*)alloc(ND * 4);
  float* HI   = (float*)alloc(ND * 4);
  float* KP   = (float*)alloc(ND * 4);
  float* PARA = (float*)alloc(ND * 4);
  unsigned short* HFB   = (unsigned short*)alloc(ND * 2);
  unsigned short* HIB   = (unsigned short*)alloc(ND * 2);
  unsigned short* PARAB = (unsigned short*)alloc(ND * 2);
  unsigned short* CATB  = (unsigned short*)alloc(ND * 2 * 2);
  float* BQ = (float*)alloc(8 * 4);

  auto zero32 = [&](void* p, size_t n32) {
    k_zero32<<<dim3((unsigned)((n32 + 255) / 256)), dim3(256), 0, stream>>>((unsigned*)p, n32);
  };
  auto conv = [&](const float* in, unsigned short* out, size_t n) {
    k_f2bf16<<<dim3((unsigned)((n + 255) / 256)), dim3(256), 0, stream>>>(in, out, n);
  };

  const int eb = (E + 255) / 256;
  const dim3 gg((unsigned)(N / 16), D_DIM / 128);

  // ---- phase 0: bf16 conversions + relation buckets ----
  conv(x, XB, ND);
  conv(WVf, WVFB, (size_t)R_REL * D_DIM * D_DIM);
  conv(WVi, WVIB, (size_t)R_REL * D_DIM * D_DIM);
  conv(Wpf, WPFB, (size_t)D_DIM * D_DIM);
  conv(Wpi, WPIB, (size_t)D_DIM * D_DIM);
  conv(Wpa, WPARAB, (size_t)2 * D_DIM * D_DIM);
  conv(Wdoc, WDOCB, (size_t)D_DIM * D_DIM);
  zero32(CNT, R_REL);
  k_bucket<<<eb, 256, 0, stream>>>(et, CNT, BUCK, E, E);

  // ---- phase 1: relation-aware GAT (both views) ----
  auto run_rgat = [&](const float* WQm, const float* WKm, const float* av,
                      const unsigned short* WVb, const float* g1, const float* b1,
                      float* Hout, unsigned short* Hb) {
    k_matvec<<<R_REL * D_DIM, 256, 0, stream>>>(WQm, av, WQv, D_DIM);
    k_matvec<<<R_REL * D_DIM, 256, 0, stream>>>(WKm, av + D_DIM, WKv, D_DIM);
    for (int r = 0; r < R_REL; ++r) {
      k_rowdot<<<N, 256, 0, stream>>>(x, D_DIM, WQv + (size_t)r * D_DIM, nullptr, SQ + r, R_REL, D_DIM);
      k_rowdot<<<N, 256, 0, stream>>>(x, D_DIM, WKv + (size_t)r * D_DIM, nullptr, SK + r, R_REL, D_DIM);
    }
    zero32(SEGMAX, (size_t)N * R_REL);
    zero32(SEGSUM, (size_t)N * R_REL);
    k_edge_logits<<<eb, 256, 0, stream>>>(tgt, src, et, R_REL, SQ, SK, nullptr, EATT, SEGMAX, E);
    k_edge_exp<<<eb, 256, 0, stream>>>(tgt, et, R_REL, EATT, SEGMAX, SEGSUM, E);
    k_edge_att<<<eb, 256, 0, stream>>>(tgt, et, R_REL, EATT, SEGSUM, E);
    zero32(ACC, ND);
    dim3 ga((unsigned)((E + 15) / 16), D_DIM / 128, R_REL);
    k_rgat_agg<<<ga, 256, 0, stream>>>(XB, WVb, BUCK, CNT, src, tgt, EATT, ACC, E);
    k_ln<<<N, 256, 0, stream>>>(x, ACC, 1, g1, b1, Hout, Hb, D_DIM, 0);
  };
  run_rgat(WQf, WKf, af, WVFB, gf1, bf1, HF, HFB);
  run_rgat(WQi, WKi, ai, WVIB, gi1, bi1, HI, HIB);

  // ---- phase 2: cross-view attention (both directions) ----
  auto run_cross = [&](const unsigned short* kvB, const float* Qf, const float* KVf,
                       const float* Wp, const unsigned short* WpB, const float* bp,
                       const float* wa, const float* ba,
                       const int* segI, const int* othI,
                       const float* baseF, const float* g2, const float* b2, int catOff) {
    k_gemm<<<gg, 256, 0, stream>>>(kvB, D_DIM, WpB, D_DIM, bp, KP, nullptr, 0, D_DIM);
    k_matvec<<<D_DIM, 256, 0, stream>>>(Wp, wa, WQv, D_DIM);
    k_matvec<<<D_DIM, 256, 0, stream>>>(Wp, wa + D_DIM, WKv, D_DIM);
    k_dot<<<1, 256, 0, stream>>>(bp, wa, BQ, D_DIM);
    k_dot<<<1, 256, 0, stream>>>(bp, wa + D_DIM, BQ + 1, D_DIM);
    k_rowdot<<<N, 256, 0, stream>>>(Qf, D_DIM, WQv, BQ, SQ, 1, D_DIM);
    k_rowdot<<<N, 256, 0, stream>>>(KVf, D_DIM, WKv, BQ + 1, SK, 1, D_DIM);
    zero32(SEGMAX, (size_t)N);
    zero32(SEGSUM, (size_t)N);
    k_edge_logits<<<eb, 256, 0, stream>>>(segI, othI, nullptr, 1, SQ, SK, ba, EATT, SEGMAX, E);
    k_edge_exp<<<eb, 256, 0, stream>>>(segI, nullptr, 1, EATT, SEGMAX, SEGSUM, E);
    k_edge_att<<<eb, 256, 0, stream>>>(segI, nullptr, 1, EATT, SEGSUM, E);
    zero32(ACC, ND);
    k_scatter<<<E, 256, 0, stream>>>(segI, othI, EATT, KP, ACC);
    k_ln<<<N, 256, 0, stream>>>(baseF, ACC, 0, g2, b2, nullptr, CATB, 2 * D_DIM, catOff);
  };
  // cf: q=hf, kv=hi, seg=qi, gather=ki
  run_cross(HIB, HF, HI, Wpf, WPFB, bpf, waf, baf, qi, ki, HF, gf2, bf2, 0);
  // ci: q=hi, kv=hf, seg=ki, gather=qi
  run_cross(HFB, HI, HF, Wpi, WPIB, bpi, wai, bai, ki, qi, HI, gi2, bi2, D_DIM);

  // ---- phase 3: para projection (concat 2D -> D) ----
  k_gemm<<<gg, 256, 0, stream>>>(CATB, 2 * D_DIM, WPARAB, D_DIM, bpa, PARA, PARAB, D_DIM, 2 * D_DIM);

  // ---- phase 4: doc GAT ----
  k_gemm<<<gg, 256, 0, stream>>>(PARAB, D_DIM, WDOCB, D_DIM, nullptr, KP, nullptr, 0, D_DIM);
  k_rowdot<<<N, 256, 0, stream>>>(KP, D_DIM, adoc, nullptr, SQ, 1, D_DIM);           // src side (a[:D])
  k_rowdot<<<N, 256, 0, stream>>>(KP, D_DIM, adoc + D_DIM, nullptr, SK, 1, D_DIM);   // tgt side (a[D:])
  zero32(SEGMAX, (size_t)N);
  zero32(SEGSUM, (size_t)N);
  k_edge_logits<<<eb, 256, 0, stream>>>(dtg, dsr, nullptr, 1, SK, SQ, nullptr, EATT, SEGMAX, E);
  k_edge_exp<<<eb, 256, 0, stream>>>(dtg, nullptr, 1, EATT, SEGMAX, SEGSUM, E);
  k_edge_att<<<eb, 256, 0, stream>>>(dtg, nullptr, 1, EATT, SEGSUM, E);
  zero32(ACC, ND);
  k_scatter<<<E, 256, 0, stream>>>(dtg, dsr, EATT, KP, ACC);
  k_ln<<<N, 256, 0, stream>>>(PARA, ACC, 1, gdoc, bdoc, (float*)d_out, nullptr, 0, 0);
}
